// DCN_59433757442261
// MI455X (gfx1250) — matrix operand order repaired and checked by reading, not compiled, not run on hardware
//
#include <hip/hip_runtime.h>
#include <hip/hip_bf16.h>
#include <math.h>

typedef __attribute__((ext_vector_type(16))) _Float16 v16h;
typedef __attribute__((ext_vector_type(8)))  float    v8f;
typedef __attribute__((ext_vector_type(2)))  _Float16 h2;

#define Bsz   4096
#define Fcat  26
#define Vocab 10000
#define Demb  64
#define D0    1664      // F*D
#define NUMB  13
#define INF_  1677      // D0 + NUMB
#define KPAD  1696      // INF_ padded to multiple of 32
#define H1    1024
#define H2    512
#define H3    256
#define EPS   1e-5f

// ---------------------------------------------------------------------------
// Weight convert + pad:  dst[n][k] = (k < Ks) ? f16(src[n][k]) : 0
// ---------------------------------------------------------------------------
__global__ void conv_pad_f16(const float* __restrict__ src, _Float16* __restrict__ dst,
                             int N, int Ks, int Kd) {
    size_t i = (size_t)blockIdx.x * blockDim.x + threadIdx.x;
    size_t total = (size_t)N * Kd;
    if (i >= total) return;
    int n = (int)(i / Kd);
    int k = (int)(i % Kd);
    dst[i] = (k < Ks) ? (_Float16)src[(size_t)n * Ks + k] : (_Float16)0.0f;
}

// ---------------------------------------------------------------------------
// Per-row: embedding gather -> LDS, write bn0(h0) as f16 (padded to KPAD),
// 3x cross iterations, and partial pred dot over x (first D0 pred weights).
// One block (256 threads) per batch row.
// ---------------------------------------------------------------------------
__global__ void embed_cross_kernel(const float* __restrict__ numb,
                                   const int*   __restrict__ cat,
                                   const float* __restrict__ emb,
                                   const float* __restrict__ bn0,
                                   const float* __restrict__ cross_w,
                                   const float* __restrict__ cross_b,
                                   const float* __restrict__ pred_w,
                                   _Float16*    __restrict__ h0,
                                   float*       __restrict__ predpart) {
    __shared__ float xrow[D0];
    __shared__ float red[256];
    __shared__ int   cats[Fcat];

    const int b = blockIdx.x;
    const int t = threadIdx.x;

    if (t < Fcat) cats[t] = cat[b * Fcat + t];
    __syncthreads();

    // gather embeddings
    for (int j = t; j < D0; j += 256) {
        int f = j >> 6;          // /64
        int d = j & 63;
        xrow[j] = emb[((size_t)f * Vocab + (size_t)cats[f]) * Demb + d];
    }
    __syncthreads();

    // h0 = f16(bn0(concat(x0, numb))), zero-padded to KPAD
    for (int j = t; j < KPAD; j += 256) {
        float y = 0.0f;
        if (j < INF_) {
            float x = (j < D0) ? xrow[j] : numb[b * NUMB + (j - D0)];
            float g  = bn0[0 * INF_ + j];
            float be = bn0[1 * INF_ + j];
            float mu = bn0[2 * INF_ + j];
            float va = bn0[3 * INF_ + j];
            y = (x - mu) * rsqrtf(va + EPS) * g + be;
        }
        h0[(size_t)b * KPAD + j] = (_Float16)y;
    }

    // cross network: x = x*(1 + dot(x, w_i)) + b_i
    for (int i = 0; i < 3; ++i) {
        float s = 0.0f;
        for (int j = t; j < D0; j += 256) s += xrow[j] * cross_w[i * D0 + j];
        red[t] = s;
        __syncthreads();
        for (int off = 128; off > 0; off >>= 1) {
            if (t < off) red[t] += red[t + off];
            __syncthreads();
        }
        float scale = 1.0f + red[0];
        float cb = cross_b[i];
        __syncthreads();
        for (int j = t; j < D0; j += 256) xrow[j] = xrow[j] * scale + cb;
        __syncthreads();
    }

    // partial prediction: dot(x, pred_w[0:D0])
    {
        float s = 0.0f;
        for (int j = t; j < D0; j += 256) s += xrow[j] * pred_w[j];
        red[t] = s;
        __syncthreads();
        for (int off = 128; off > 0; off >>= 1) {
            if (t < off) red[t] += red[t + off];
            __syncthreads();
        }
        if (t == 0) predpart[b] = red[0];
    }
}

// ---------------------------------------------------------------------------
// WMMA GEMM:  C[M,N] = A[M,K] @ W[N,K]^T, epilogue bias + batchnorm.
// A row-major f16 (K contiguous), W row-major [N][K] f16.
// Register-blocked: each wave computes a 32(M) x 64(N) tile:
//   2 A-frags x 4 B-frags -> 8 WMMAs / K-step, 8 v8f accumulators.
// Block = 256 threads = 8 waves arranged 4(M) x 2(N): 128 x 128 block tile.
// grid = (N/128, M/128).
// ---------------------------------------------------------------------------
__global__ void wmma_gemm_bn(const _Float16* __restrict__ A,
                             const _Float16* __restrict__ W,
                             const float*    __restrict__ bias,
                             const float*    __restrict__ bn,     // (4, N)
                             _Float16*       __restrict__ outF16, // one of these null
                             float*          __restrict__ outF32,
                             int M, int N, int K) {
    const int lane = threadIdx.x & 31;
    const int wave = threadIdx.x >> 5;
    const int wm   = wave & 3;     // M sub-tile within block
    const int wn   = wave >> 2;    // N sub-tile within block
    const int half = lane >> 4;    // 0: lanes 0-15, 1: lanes 16-31
    const int l16  = lane & 15;

    const int m0 = blockIdx.y * 128 + wm * 32;  // two 16-row tiles: m0, m0+16
    const int n0 = blockIdx.x * 128 + wn * 64;  // four 16-col tiles

    const _Float16* Arow0 = A + (size_t)(m0 + l16) * K;
    const _Float16* Arow1 = A + (size_t)(m0 + 16 + l16) * K;
    const _Float16* Wr0 = W + (size_t)(n0 +  0 + l16) * K;
    const _Float16* Wr1 = W + (size_t)(n0 + 16 + l16) * K;
    const _Float16* Wr2 = W + (size_t)(n0 + 32 + l16) * K;
    const _Float16* Wr3 = W + (size_t)(n0 + 48 + l16) * K;

    v8f acc[2][4] = {};
    for (int k0 = 0; k0 < K; k0 += 32) {
        v16h a0, a1, bf0, bf1, bf2, bf3;
#pragma unroll
        for (int v = 0; v < 8; ++v) {
            // A 16x32 fragment: f16 pairs at k = 2*(v&3) + 16*(v>>2) + 8*half
            int ka = k0 + 2 * (v & 3) + 16 * (v >> 2) + 8 * half;
            h2 p0 = *(const h2*)(Arow0 + ka);
            h2 p1 = *(const h2*)(Arow1 + ka);
            a0[2 * v] = p0.x; a0[2 * v + 1] = p0.y;
            a1[2 * v] = p1.x; a1[2 * v + 1] = p1.y;
            // B 32x16 fragment: f16 pairs at k = 2*(v&3) + 16*(v>>2) + 8*half
            int kb = k0 + 2 * (v & 3) + 16 * (v >> 2) + 8 * half;
            h2 q0 = *(const h2*)(Wr0 + kb);
            h2 q1 = *(const h2*)(Wr1 + kb);
            h2 q2 = *(const h2*)(Wr2 + kb);
            h2 q3 = *(const h2*)(Wr3 + kb);
            bf0[2 * v] = q0.x; bf0[2 * v + 1] = q0.y;
            bf1[2 * v] = q1.x; bf1[2 * v + 1] = q1.y;
            bf2[2 * v] = q2.x; bf2[2 * v + 1] = q2.y;
            bf3[2 * v] = q3.x; bf3[2 * v + 1] = q3.y;
        }
        acc[0][0] = __builtin_amdgcn_wmma_f32_16x16x32_f16(false, a0, false, bf0, (short)0, acc[0][0], false, false);
        acc[1][0] = __builtin_amdgcn_wmma_f32_16x16x32_f16(false, a1, false, bf0, (short)0, acc[1][0], false, false);
        acc[0][1] = __builtin_amdgcn_wmma_f32_16x16x32_f16(false, a0, false, bf1, (short)0, acc[0][1], false, false);
        acc[1][1] = __builtin_amdgcn_wmma_f32_16x16x32_f16(false, a1, false, bf1, (short)0, acc[1][1], false, false);
        acc[0][2] = __builtin_amdgcn_wmma_f32_16x16x32_f16(false, a0, false, bf2, (short)0, acc[0][2], false, false);
        acc[1][2] = __builtin_amdgcn_wmma_f32_16x16x32_f16(false, a1, false, bf2, (short)0, acc[1][2], false, false);
        acc[0][3] = __builtin_amdgcn_wmma_f32_16x16x32_f16(false, a0, false, bf3, (short)0, acc[0][3], false, false);
        acc[1][3] = __builtin_amdgcn_wmma_f32_16x16x32_f16(false, a1, false, bf3, (short)0, acc[1][3], false, false);
    }

    // epilogue: y = (acc + bias[n] - mu[n]) * rsqrt(var[n]+eps) * g[n] + beta[n]
#pragma unroll
    for (int j = 0; j < 4; ++j) {
        const int n = n0 + j * 16 + l16;
        const float g  = bn[0 * N + n];
        const float be = bn[1 * N + n];
        const float mu = bn[2 * N + n];
        const float va = bn[3 * N + n];
        const float sc = rsqrtf(va + EPS) * g;
        const float bs = bias[n];
#pragma unroll
        for (int i = 0; i < 2; ++i) {
#pragma unroll
            for (int v = 0; v < 8; ++v) {
                int m = m0 + i * 16 + v + 8 * half;
                float y = (acc[i][j][v] + bs - mu) * sc + be;
                if (outF16) outF16[(size_t)m * N + n] = (_Float16)y;
                else        outF32[(size_t)m * N + n] = y;
            }
        }
    }
}

// ---------------------------------------------------------------------------
// Final: out[b] = sigmoid(predpart[b] + dot(h3[b], pred_w[D0:]) + pred_b)
// One block (256 threads) per row; H3 == 256.
// ---------------------------------------------------------------------------
__global__ void pred_kernel(const float* __restrict__ h3,
                            const float* __restrict__ predpart,
                            const float* __restrict__ pred_w,
                            const float* __restrict__ pred_b,
                            float* __restrict__ out) {
    __shared__ float red[256];
    const int b = blockIdx.x;
    const int t = threadIdx.x;
    red[t] = h3[(size_t)b * H3 + t] * pred_w[D0 + t];
    __syncthreads();
    for (int off = 128; off > 0; off >>= 1) {
        if (t < off) red[t] += red[t + off];
        __syncthreads();
    }
    if (t == 0) {
        float z = predpart[b] + red[0] + pred_b[0];
        out[b] = 1.0f / (1.0f + expf(-z));
    }
}

// ---------------------------------------------------------------------------
extern "C" void kernel_launch(void* const* d_in, const int* in_sizes, int n_in,
                              void* d_out, int out_size, void* d_ws, size_t ws_size,
                              hipStream_t stream) {
    const float* numb    = (const float*)d_in[0];
    const int*   cat     = (const int*)  d_in[1];
    const float* emb     = (const float*)d_in[2];
    const float* bn0     = (const float*)d_in[3];
    const float* w1      = (const float*)d_in[4];
    const float* b1      = (const float*)d_in[5];
    const float* bn1     = (const float*)d_in[6];
    const float* w2      = (const float*)d_in[7];
    const float* b2      = (const float*)d_in[8];
    const float* bn2     = (const float*)d_in[9];
    const float* w3      = (const float*)d_in[10];
    const float* b3      = (const float*)d_in[11];
    const float* bn3     = (const float*)d_in[12];
    const float* cross_w = (const float*)d_in[13];
    const float* cross_b = (const float*)d_in[14];
    const float* pred_w  = (const float*)d_in[15];
    const float* pred_b  = (const float*)d_in[16];
    float* out = (float*)d_out;

    // workspace carve-up
    char* ws = (char*)d_ws;
    size_t off = 0;
    _Float16* h0f = (_Float16*)(ws + off); off += (size_t)Bsz * KPAD * 2;  // 13.9 MB
    _Float16* w1f = (_Float16*)(ws + off); off += (size_t)H1  * KPAD * 2;  //  3.5 MB
    _Float16* h1f = (_Float16*)(ws + off); off += (size_t)Bsz * H1   * 2;  //  8.4 MB
    _Float16* w2f = (_Float16*)(ws + off); off += (size_t)H2  * H1   * 2;  //  1.0 MB
    _Float16* h2f = (_Float16*)(ws + off); off += (size_t)Bsz * H2   * 2;  //  4.2 MB
    _Float16* w3f = (_Float16*)(ws + off); off += (size_t)H3  * H2   * 2;  //  0.3 MB
    float*    h3f = (float*)   (ws + off); off += (size_t)Bsz * H3   * 4;  //  4.2 MB
    float*    pp  = (float*)   (ws + off); off += (size_t)Bsz * 4;
    (void)ws_size; (void)off; (void)n_in; (void)in_sizes; (void)out_size;

    // 1) weight convert + pad to f16
    {
        size_t tot1 = (size_t)H1 * KPAD;
        conv_pad_f16<<<dim3((unsigned)((tot1 + 255) / 256)), 256, 0, stream>>>(w1, w1f, H1, INF_, KPAD);
        size_t tot2 = (size_t)H2 * H1;
        conv_pad_f16<<<dim3((unsigned)((tot2 + 255) / 256)), 256, 0, stream>>>(w2, w2f, H2, H1, H1);
        size_t tot3 = (size_t)H3 * H2;
        conv_pad_f16<<<dim3((unsigned)((tot3 + 255) / 256)), 256, 0, stream>>>(w3, w3f, H3, H2, H2);
    }

    // 2) embedding gather + cross + bn0 pack + partial pred dot
    embed_cross_kernel<<<Bsz, 256, 0, stream>>>(numb, cat, emb, bn0, cross_w, cross_b,
                                                pred_w, h0f, pp);

    // 3) GEMM1: (B,KPAD) @ (H1,KPAD)^T -> bn1 -> h1 f16
    wmma_gemm_bn<<<dim3(H1 / 128, Bsz / 128), 256, 0, stream>>>(
        h0f, w1f, b1, bn1, h1f, nullptr, Bsz, H1, KPAD);

    // 4) GEMM2: (B,H1) @ (H2,H1)^T -> bn2 -> h2 f16
    wmma_gemm_bn<<<dim3(H2 / 128, Bsz / 128), 256, 0, stream>>>(
        h1f, w2f, b2, bn2, h2f, nullptr, Bsz, H2, H1);

    // 5) GEMM3: (B,H2) @ (H3,H2)^T -> bn3 -> h3 f32
    wmma_gemm_bn<<<dim3(H3 / 128, Bsz / 128), 256, 0, stream>>>(
        h2f, w3f, b3, bn3, nullptr, h3f, Bsz, H3, H2);

    // 6) final dot + sigmoid
    pred_kernel<<<Bsz, 256, 0, stream>>>(h3f, pp, pred_w, pred_b, out);
}
